// GRAM_68831145886256
// MI455X (gfx1250) — compile-verified, matrix operand order checked by reading, not run on hardware
//
#include <hip/hip_runtime.h>
#include <math.h>
#include <stdint.h>

// ---------------------------------------------------------------------------
// CDNA5 / gfx1250 WMMA types and helpers (wave32)
// ---------------------------------------------------------------------------
typedef __attribute__((ext_vector_type(16))) __bf16 v16bf;
typedef __attribute__((ext_vector_type(8)))  float  v8f;

#define DEV_INLINE __device__ __forceinline__

DEV_INLINE unsigned short f32_to_bf16_rne(float f) {
  unsigned int u = __float_as_uint(f);
  u += 0x7FFFu + ((u >> 16) & 1u);
  return (unsigned short)(u >> 16);
}

union FragU {
  v16bf v;
  unsigned short u[16];
  uint4 q[2];
};

// 16x32 bf16 fragment at (row0, k0) of a row-major [*, lda] ushort matrix.
// ISA layout (16-bit A 16x32): lane&15 = row; element e ->
//   K = (e>>3)*16 + (lane>>4)*8 + (e&7)
// The two 8-element groups are contiguous in K -> exactly two b128 loads per
// lane, NO guards: callers guarantee lda (=K) is a multiple of 32 (padded).
// Used for BOTH operands: A is [M,K]; B is stored transposed as Bt[N,K].
DEV_INLINE v16bf load_frag(const unsigned short* P, int lda, int row0, int k0) {
  const int lane = threadIdx.x & 31;
  const int r    = lane & 15;
  const int half = lane >> 4;
  FragU f;
  const unsigned short* p = P + (size_t)(row0 + r) * lda + k0 + half * 8;
  f.q[0] = *(const uint4*)(p);
  f.q[1] = *(const uint4*)(p + 16);
  return f.v;
}

// ---------------------------------------------------------------------------
// Generic WMMA GEMM with transposed B:
//   C[M,N](f32) = act(A_bf16[M,K] @ Bt_bf16[N,K]^T + bias) * rowscale
// K must be a multiple of 32 (zero-padded). Bt padded to ceil16(N) rows.
// Block = 256 threads = 8 waves; tile = 16 rows x 128 cols; wave = 16 cols.
// act: 0 none, 1 tanh, 2 sigmoid. C and Cbf each optional.
// ---------------------------------------------------------------------------
__global__ void gemm_bt_wmma(const unsigned short* __restrict__ A,
                             const unsigned short* __restrict__ Bt,
                             float* __restrict__ C,
                             unsigned short* __restrict__ Cbf,
                             const float* __restrict__ bias,
                             const float* __restrict__ rowscale,
                             int M, int N, int K, int act) {
  const int wave = threadIdx.x >> 5;
  const int lane = threadIdx.x & 31;
  const int row0 = blockIdx.x * 16;
  const int col0 = (blockIdx.y * 8 + wave) * 16;
  if (col0 >= N) return;                 // wave-uniform: EXEC stays all-ones
  v8f acc = {};
  for (int k0 = 0; k0 < K; k0 += 32) {
    __builtin_prefetch(A + (size_t)(row0 + (lane & 15)) * K + k0 + 256, 0, 1);
    v16bf a = load_frag(A,  K, row0, k0);
    v16bf b = load_frag(Bt, K, col0, k0);
    acc = __builtin_amdgcn_wmma_f32_16x16x32_bf16(false, a, false, b,
                                                  (short)0, acc, false, false);
  }
  const int n = col0 + (lane & 15);
  if (n < N) {
#pragma unroll
    for (int v = 0; v < 8; ++v) {
      int m = row0 + v + ((lane >> 4) << 3);
      float val = acc[v];
      if (bias) val += bias[n];
      if (act == 1) val = tanhf(val);
      else if (act == 2) val = 1.0f / (1.0f + __expf(-val));
      if (rowscale) val *= rowscale[m];
      size_t o = (size_t)m * N + n;
      if (C)   C[o]   = val;
      if (Cbf) Cbf[o] = f32_to_bf16_rne(val);
    }
  }
}

// ---------------------------------------------------------------------------
// DAG attention: 2 concepts x 8 ancestors = one 16-row WMMA tile per block.
// The gathered [leaf|anc] tile is staged into LDS with CDNA5 async global->LDS
// DMA (global_load_async_to_lds_b128, tracked by ASYNCcnt).
// hidden = tanh(xcat @ W1 + b1), score = hidden @ W2 + b2, masked softmax,
// dag_emb = sum_a aw * anc_emb.  Writes dag_emb TRANSPOSED [E, Vpad] in bf16.
// ---------------------------------------------------------------------------
__global__ void dag_attn_wmma(const int* __restrict__ leaves,
                              const int* __restrict__ ancestors,
                              const float* __restrict__ masksl,
                              const unsigned short* __restrict__ embed_bf,
                              const float* __restrict__ embed_f32,
                              const unsigned short* __restrict__ W1T, // [128,256]
                              const float* __restrict__ b1,
                              const float* __restrict__ W2,           // [128]
                              const float* __restrict__ b2,
                              unsigned short* __restrict__ dag_bfT,   // [E, Vpad]
                              int Vpad) {
  const int AA = 8, EE = 128;
  __shared__ unsigned short xcat[16][256];  // 16 rows x (leaf 128 | anc 128)
  __shared__ float hid[16][128];
  __shared__ float sc[16];
  __shared__ float aw[16];
  __shared__ int   aidx_s[16];
  const int v0  = blockIdx.x * 2;
  const int tid = threadIdx.x;

  { // async-stage xcat: each thread DMAs two 16-byte chunks global->LDS
    int r = tid >> 4;                   // 0..15 (16 threads per row)
    int v = v0 + (r >> 3);
    int a = r & 7;
    int li = leaves[v * AA + a];
    int ai = ancestors[v * AA + a];
    if ((tid & 15) == 0) aidx_s[r] = ai;
    const unsigned short* lrow = embed_bf + (size_t)li * EE;
    const unsigned short* arow = embed_bf + (size_t)ai * EE;
#pragma unroll
    for (int c = 0; c < 2; ++c) {
      int seg = (tid & 15) * 2 + c;     // 0..31, 8 ushorts each
      const unsigned short* src = (seg < 16) ? (lrow + seg * 8)
                                             : (arow + (seg - 16) * 8);
      unsigned int lds_off = (unsigned int)(uintptr_t)(&xcat[r][seg * 8]);
      unsigned long long ga = (unsigned long long)(uintptr_t)src;
      asm volatile("global_load_async_to_lds_b128 %0, %1, off"
                   :: "v"(lds_off), "v"(ga) : "memory");
    }
    asm volatile("s_wait_asynccnt 0x0" ::: "memory");
  }
  __syncthreads();

  { // hidden = tanh(xcat @ W1 + b1) via WMMA, wave w -> cols 16w..16w+15
    const int lane = tid & 31;
    const int col0 = (tid >> 5) * 16;
    v8f acc = {};
    for (int k0 = 0; k0 < 256; k0 += 32) {
      v16bf a = load_frag(&xcat[0][0], 256, 0, k0);
      v16bf b = load_frag(W1T, 256, col0, k0);
      acc = __builtin_amdgcn_wmma_f32_16x16x32_bf16(false, a, false, b,
                                                    (short)0, acc, false, false);
    }
    const int n = col0 + (lane & 15);
#pragma unroll
    for (int v = 0; v < 8; ++v) {
      int m = v + ((lane >> 4) << 3);
      hid[m][n] = tanhf(acc[v] + b1[n]);
    }
  }
  __syncthreads();

  if (tid < 16) { // scores
    float s = b2[0];
#pragma unroll 4
    for (int j = 0; j < 128; ++j) s += hid[tid][j] * W2[j];
    int v = v0 + (tid >> 3), a = tid & 7;
    s += (1.0f - masksl[v * AA + a]) * -10000.0f;
    sc[tid] = s;
  }
  __syncthreads();

  if (tid < 2) { // softmax over the 8 ancestors of each concept
    float mx = -1e30f;
    for (int a = 0; a < 8; ++a) mx = fmaxf(mx, sc[tid * 8 + a]);
    float e[8], sum = 0.f;
    for (int a = 0; a < 8; ++a) { e[a] = __expf(sc[tid * 8 + a] - mx); sum += e[a]; }
    for (int a = 0; a < 8; ++a) aw[tid * 8 + a] = e[a] / sum;
  }
  __syncthreads();

  { // dag_embT[e, v] = sum_a aw * anc_emb (f32 gather for accuracy)
    int vloc = tid >> 7;       // 0..1
    int ecol = tid & 127;
    float s = 0.f;
#pragma unroll
    for (int a = 0; a < 8; ++a)
      s += aw[vloc * 8 + a] * embed_f32[(size_t)aidx_s[vloc * 8 + a] * EE + ecol];
    dag_bfT[(size_t)ecol * Vpad + (v0 + vloc)] = f32_to_bf16_rne(s);
  }
}

// ---------------------------------------------------------------------------
// GRU scan: single persistent workgroup (32 waves), h + gh resident in LDS
// (295 KB dynamic LDS -- CDNA5 WGP has 320 KB). Per step:
//   gh = h @ Wh^T (WMMA with Wh already [3H, H] = Bt layout), fused gate math.
// ---------------------------------------------------------------------------
__global__ void gru_scan_wmma(const float* __restrict__ xp,        // [B*T, 3H]
                              const unsigned short* __restrict__ Whbf, // [3H, H]
                              const float* __restrict__ bh,        // [3H]
                              float* __restrict__ hs,              // [B*T, H]
                              unsigned short* __restrict__ hs_bf,
                              int Bn, int T, int H) {
  extern __shared__ unsigned char smem_raw[];
  float*          h   = (float*)smem_raw;                 // [Bn*H]  f32
  unsigned short* hbf = (unsigned short*)(h + Bn * H);    // [Bn*H]  bf16 mirror
  float*          gh  = (float*)(hbf + Bn * H);           // [Bn*3H] f32
  const int N3 = 3 * H;

  for (int i = threadIdx.x; i < Bn * H; i += blockDim.x) { h[i] = 0.f; hbf[i] = 0; }
  __syncthreads();

  const int wave   = threadIdx.x >> 5;
  const int lane   = threadIdx.x & 31;
  const int nwaves = blockDim.x >> 5;
  const int mtiles = Bn >> 4;       // 4
  const int ntiles = N3 >> 4;       // 48

  for (int t = 0; t < T; ++t) {
    for (int tile = wave; tile < mtiles * ntiles; tile += nwaves) {
      int mt = tile / ntiles, nt = tile - mt * ntiles;
      v8f acc = {};
      for (int k0 = 0; k0 < H; k0 += 32) {
        v16bf a = load_frag(hbf, H, mt * 16, k0);         // LDS b128 path
        v16bf b = load_frag(Whbf, H, nt * 16, k0);        // global b128 path
        acc = __builtin_amdgcn_wmma_f32_16x16x32_bf16(false, a, false, b,
                                                      (short)0, acc, false, false);
      }
      int n = nt * 16 + (lane & 15);
#pragma unroll
      for (int v = 0; v < 8; ++v) {
        int m = mt * 16 + v + ((lane >> 4) << 3);
        gh[m * N3 + n] = acc[v];
      }
    }
    __syncthreads();
    for (int i = threadIdx.x; i < Bn * H; i += blockDim.x) {
      int b = i / H, j = i - b * H;
      const float* xrow = xp + ((size_t)b * T + t) * N3;
      float hr = gh[b * N3 + j]         + bh[j];
      float hz = gh[b * N3 + H + j]     + bh[H + j];
      float hn = gh[b * N3 + 2 * H + j] + bh[2 * H + j];
      float r  = 1.0f / (1.0f + __expf(-(xrow[j] + hr)));
      float z  = 1.0f / (1.0f + __expf(-(xrow[H + j] + hz)));
      float nn = tanhf(xrow[2 * H + j] + r * hn);
      float hnew = (1.0f - z) * nn + z * h[i];
      h[i]   = hnew;
      hbf[i] = f32_to_bf16_rne(hnew);
      size_t o = ((size_t)b * T + t) * H + j;
      hs[o] = hnew;
      hs_bf[o] = f32_to_bf16_rne(hnew);
    }
    __syncthreads();
  }
}

// ---------------------------------------------------------------------------
// Elementwise converts + fills + final gather
// ---------------------------------------------------------------------------
__global__ void cvt_f32_bf16(const float* __restrict__ src,
                             unsigned short* __restrict__ dst, int n) {
  int i = blockIdx.x * blockDim.x + threadIdx.x;
  if (i < n) dst[i] = f32_to_bf16_rne(src[i]);
}

// src [M, K] f32 -> dst [M, Kpad] bf16, columns >= K zero-filled
__global__ void cvt_f32_bf16_padK(const float* __restrict__ src,
                                  unsigned short* __restrict__ dst,
                                  int M, int K, int Kpad) {
  int i = blockIdx.x * blockDim.x + threadIdx.x;
  if (i < M * Kpad) {
    int m = i / Kpad, k = i - m * Kpad;
    dst[i] = (k < K) ? f32_to_bf16_rne(src[(size_t)m * K + k]) : (unsigned short)0;
  }
}

// src [K, N] f32 -> dst [Npad, K] bf16 (transposed, rows >= N zero-filled)
__global__ void cvt_bt_pad(const float* __restrict__ src,
                           unsigned short* __restrict__ dst,
                           int K, int N, int Npad) {
  int i = blockIdx.x * blockDim.x + threadIdx.x;
  if (i < Npad * K) {
    int n = i / K, k = i - n * K;
    dst[i] = (n < N) ? f32_to_bf16_rne(src[(size_t)k * N + n]) : (unsigned short)0;
  }
}

__global__ void fill_zero_ush(unsigned short* __restrict__ dst, int n) {
  int i = blockIdx.x * blockDim.x + threadIdx.x;
  if (i < n) dst[i] = 0;
}

__global__ void last_gather(const float* __restrict__ out,
                            const int* __restrict__ lengths,
                            float* __restrict__ last, int Bn, int T, int C) {
  int i = blockIdx.x * blockDim.x + threadIdx.x;
  if (i < Bn * C) {
    int b = i / C, c = i - b * C;
    int idx = lengths[b];
    idx = idx < 1 ? 1 : (idx > T ? T : idx);
    idx -= 1;
    last[i] = out[((size_t)b * T + idx) * C + c];
  }
}

// ---------------------------------------------------------------------------
// Host orchestration
// ---------------------------------------------------------------------------
extern "C" void kernel_launch(void* const* d_in, const int* in_sizes, int n_in,
                              void* d_out, int out_size, void* d_ws, size_t ws_size,
                              hipStream_t stream) {
  const int B = 64, T = 48, V = 10000, E = 128, H = 256, C = 283, NC = 11200;
  const int BT = B * T;            // 3072
  const int VP = 10016;            // V padded to a multiple of 32
  const int CP = 288;              // C padded to 16
  (void)in_sizes; (void)n_in; (void)out_size; (void)ws_size;

  const float* inputs  = (const float*)d_in[0];
  const float* mask    = (const float*)d_in[1];
  const int*   lengths = (const int*)d_in[2];
  const int*   leaves  = (const int*)d_in[3];
  const int*   anc     = (const int*)d_in[4];
  const float* masksl  = (const float*)d_in[5];
  const float* embed   = (const float*)d_in[6];
  const float* W1      = (const float*)d_in[7];
  const float* b1      = (const float*)d_in[8];
  const float* W2      = (const float*)d_in[9];
  const float* b2      = (const float*)d_in[10];
  const float* Wx0     = (const float*)d_in[11];
  const float* Wh0     = (const float*)d_in[12];
  const float* bx0     = (const float*)d_in[13];
  const float* bh0     = (const float*)d_in[14];
  const float* Wx1     = (const float*)d_in[15];
  const float* Wh1     = (const float*)d_in[16];
  const float* bx1     = (const float*)d_in[17];
  const float* bh1     = (const float*)d_in[18];
  const float* Wf      = (const float*)d_in[19];
  const float* bfb     = (const float*)d_in[20];

  // workspace carve-up (~98 MB)
  uintptr_t p = (uintptr_t)d_ws;
  auto wsalloc = [&](size_t bytes) {
    uintptr_t r = (p + 255) & ~(uintptr_t)255;
    p = r + bytes;
    return (void*)r;
  };
  unsigned short* inputs_bf = (unsigned short*)wsalloc((size_t)BT * VP * 2);
  unsigned short* embed_bf  = (unsigned short*)wsalloc((size_t)NC * E * 2);
  unsigned short* W1T_bf    = (unsigned short*)wsalloc((size_t)E * 2 * E * 2);
  unsigned short* dag_bfT   = (unsigned short*)wsalloc((size_t)E * VP * 2);
  unsigned short* x_bf      = (unsigned short*)wsalloc((size_t)BT * E * 2);
  unsigned short* Wx0_bf    = (unsigned short*)wsalloc((size_t)3 * H * E * 2);
  unsigned short* Wh0_bf    = (unsigned short*)wsalloc((size_t)3 * H * H * 2);
  unsigned short* Wx1_bf    = (unsigned short*)wsalloc((size_t)3 * H * H * 2);
  unsigned short* Wh1_bf    = (unsigned short*)wsalloc((size_t)3 * H * H * 2);
  unsigned short* WfT_bf    = (unsigned short*)wsalloc((size_t)CP * H * 2);
  float*          xp0       = (float*)wsalloc((size_t)BT * 3 * H * 4);
  float*          h1_f32    = (float*)wsalloc((size_t)BT * H * 4);
  unsigned short* h1_bf     = (unsigned short*)wsalloc((size_t)BT * H * 2);
  float*          xp1       = (float*)wsalloc((size_t)BT * 3 * H * 4);
  float*          h2_f32    = (float*)wsalloc((size_t)BT * H * 4);
  unsigned short* h2_bf     = (unsigned short*)wsalloc((size_t)BT * H * 2);

  const int TPB = 256;
  auto cdiv = [](int a, int b) { return (a + b - 1) / b; };

  // ---- precision converts (Wx*/Wh* are already [N,K] = Bt layout) ----
  cvt_f32_bf16_padK<<<cdiv(BT * VP, TPB), TPB, 0, stream>>>(inputs, inputs_bf,
                                                            BT, V, VP);
  cvt_f32_bf16<<<cdiv(NC * E, TPB), TPB, 0, stream>>>(embed, embed_bf, NC * E);
  cvt_bt_pad<<<cdiv(E * 2 * E, TPB), TPB, 0, stream>>>(W1, W1T_bf, 2 * E, E, E);
  cvt_f32_bf16<<<cdiv(3 * H * E, TPB), TPB, 0, stream>>>(Wx0, Wx0_bf, 3 * H * E);
  cvt_f32_bf16<<<cdiv(3 * H * H, TPB), TPB, 0, stream>>>(Wh0, Wh0_bf, 3 * H * H);
  cvt_f32_bf16<<<cdiv(3 * H * H, TPB), TPB, 0, stream>>>(Wx1, Wx1_bf, 3 * H * H);
  cvt_f32_bf16<<<cdiv(3 * H * H, TPB), TPB, 0, stream>>>(Wh1, Wh1_bf, 3 * H * H);
  cvt_bt_pad<<<cdiv(CP * H, TPB), TPB, 0, stream>>>(Wf, WfT_bf, H, C, CP);
  fill_zero_ush<<<cdiv(E * VP, TPB), TPB, 0, stream>>>(dag_bfT, E * VP);

  // ---- DAG attention (async LDS staging + WMMA); fills dag_bfT[:, 0:V) ----
  dag_attn_wmma<<<V / 2, 256, 0, stream>>>(leaves, anc, masksl, embed_bf, embed,
                                           W1T_bf, b1, W2, b2, dag_bfT, VP);

  // ---- x = tanh(inputs @ dag_emb)  [3072 x 10016] x [10016 x 128] ----
  gemm_bt_wmma<<<dim3(BT / 16, 1), 256, 0, stream>>>(
      inputs_bf, dag_bfT, nullptr, x_bf, nullptr, nullptr, BT, E, VP, 1);

  // ---- GRU layer 0 ----
  gemm_bt_wmma<<<dim3(BT / 16, cdiv(3 * H, 128)), 256, 0, stream>>>(
      x_bf, Wx0_bf, xp0, nullptr, bx0, nullptr, BT, 3 * H, E, 0);
  const int scan_smem = B * H * 4 + B * H * 2 + B * 3 * H * 4;  // 294912 B < 320 KB
  (void)hipFuncSetAttribute((const void*)gru_scan_wmma,
                            hipFuncAttributeMaxDynamicSharedMemorySize, scan_smem);
  gru_scan_wmma<<<1, 1024, scan_smem, stream>>>(xp0, Wh0_bf, bh0, h1_f32, h1_bf, B, T, H);

  // ---- GRU layer 1 ----
  gemm_bt_wmma<<<dim3(BT / 16, cdiv(3 * H, 128)), 256, 0, stream>>>(
      h1_bf, Wx1_bf, xp1, nullptr, bx1, nullptr, BT, 3 * H, H, 0);
  gru_scan_wmma<<<1, 1024, scan_smem, stream>>>(xp1, Wh1_bf, bh1, h2_f32, h2_bf, B, T, H);

  // ---- out = sigmoid(h2 @ Wf + bf) * mask  -> d_out[0 .. BT*C) ----
  gemm_bt_wmma<<<dim3(BT / 16, cdiv(C, 128)), 256, 0, stream>>>(
      h2_bf, WfT_bf, (float*)d_out, nullptr, bfb, mask, BT, C, H, 2);

  // ---- last = out[b, len-1, :]  -> d_out[BT*C .. BT*C + B*C) ----
  last_gather<<<cdiv(B * C, TPB), TPB, 0, stream>>>(
      (float*)d_out, lengths, (float*)d_out + (size_t)BT * C, B, T, C);
}
